// DGN_62088047231107
// MI455X (gfx1250) — matrix operand order repaired and checked by reading, not compiled, not run on hardware
//
#include <hip/hip_runtime.h>
#include <hip/hip_bf16.h>

typedef __attribute__((ext_vector_type(16))) _Float16 v16h;
typedef __attribute__((ext_vector_type(8)))  _Float16 v8h;
typedef __attribute__((ext_vector_type(8)))  float    v8f;
typedef __attribute__((ext_vector_type(4)))  float    v4f;
typedef _Float16 half_t;

#define BATCH 128
#define NAG   512
#define DIN   64
#define HDIM  128
#define NPASS 2
#define AOUT  16

// ---------------------------------------------------------------------------
// CDNA5 async global->LDS copy (ASYNCcnt tracked), 16 bytes per lane.
// ---------------------------------------------------------------------------
__device__ __forceinline__ void async_copy_b128(const half_t* lds_dst, const half_t* gsrc) {
    unsigned int       loff  = (unsigned int)(unsigned long long)(const void*)lds_dst;
    unsigned long long gaddr = (unsigned long long)(const void*)gsrc;
    asm volatile("global_load_async_to_lds_b128 %0, %1, off"
                 :: "v"(loff), "v"(gaddr) : "memory");
}
__device__ __forceinline__ void async_wait0() {
    asm volatile("s_wait_asynccnt 0x0" ::: "memory");
}

// ---------------------------------------------------------------------------
// WMMA operand loaders (wave32, 16x16x32 f16, contiguous k = 16*hf + s order)
// ---------------------------------------------------------------------------
__device__ __forceinline__ v16h load_A_vec(const half_t* __restrict__ A,
                                           int ldA, int k0, int lane) {
    int row = lane & 15;
    int hf  = lane >> 4;
    const half_t* rp = A + (size_t)row * ldA + k0 + hf * 16;
    v8h lo = *(const v8h*)rp;
    v8h hi = *(const v8h*)(rp + 8);
    return __builtin_shufflevector(lo, hi, 0,1,2,3,4,5,6,7,8,9,10,11,12,13,14,15);
}

__device__ __forceinline__ v16h load_Bt_vec(const half_t* __restrict__ Mm,
                                            int ld, int k0, int n0, int lane) {
    int col = lane & 15;
    int hf  = lane >> 4;
    const half_t* p = Mm + (size_t)(n0 + col) * ld + k0 + hf * 16;
    v8h lo = *(const v8h*)p;
    v8h hi = *(const v8h*)(p + 8);
    return __builtin_shufflevector(lo, hi, 0,1,2,3,4,5,6,7,8,9,10,11,12,13,14,15);
}

// ---------------------------------------------------------------------------
// f32 -> f16 conversion, 8 elements per thread (b128 in, b128 out)
// ---------------------------------------------------------------------------
__global__ void cvt_f32_f16_v8(const float* __restrict__ in, half_t* __restrict__ out,
                               int n8) {
    int i = blockIdx.x * blockDim.x + threadIdx.x;
    if (i < n8) {
        const v4f* p = (const v4f*)(in + (size_t)i * 8);
        v4f a = p[0], b = p[1];
        v8h o;
#pragma unroll
        for (int t = 0; t < 4; ++t) { o[t] = (half_t)a[t]; o[4 + t] = (half_t)b[t]; }
        *(v8h*)(out + (size_t)i * 8) = o;
    }
}

// ---------------------------------------------------------------------------
// GEMM: C = [relu](A_f16[M,K] @ W_f16[K,N] + bias_f32[N])
// Block = 8 waves, covers 128 rows x N cols. W async-staged to LDS once;
// each wave pins its B regs (K/32 chunks) and sweeps row tiles.
// ---------------------------------------------------------------------------
template <int K, int N, bool RELU, bool F16OUT>
__global__ void gemm_wmma(const half_t* __restrict__ A, const half_t* __restrict__ W,
                          const float* __restrict__ bias, void* __restrict__ Cout) {
    constexpr int CT  = N / 16;
    constexpr int WPC = 8 / CT;
    constexpr int RTW = 8 / WPC;
    constexpr int NCH = K / 32;

    __shared__ __align__(16) half_t ldsW[K * N];

    int tid  = threadIdx.x;
    int lane = tid & 31;
    int wave = tid >> 5;

    constexpr int NV = (K * N) / 8;
#pragma unroll
    for (int i = tid; i < NV; i += 256)
        async_copy_b128(&ldsW[i * 8], W + i * 8);
    async_wait0();
    __syncthreads();

    int ct  = wave % CT;
    int col = lane & 15;
    int hf  = lane >> 4;

    v16h breg[NCH];
#pragma unroll
    for (int c = 0; c < NCH; ++c) {
        v16h b;
#pragma unroll
        for (int i = 0; i < 8; ++i) {
            int kk = c * 32 + hf * 16 + 2 * i;
            b[2 * i]     = ldsW[kk * N + ct * 16 + col];
            b[2 * i + 1] = ldsW[(kk + 1) * N + ct * 16 + col];
        }
        breg[c] = b;
    }

    int   n    = ct * 16 + col;
    float bval = bias[n];

#pragma unroll
    for (int j = 0; j < RTW; ++j) {
        int mt = blockIdx.x * 8 + (wave / CT) + j * WPC;
        const half_t* Arow = A + (size_t)mt * 16 * K;
        v8f acc = {};
#pragma unroll
        for (int c = 0; c < NCH; ++c) {
            v16h a = load_A_vec(Arow, K, c * 32, lane);
            acc = __builtin_amdgcn_wmma_f32_16x16x32_f16(false, a, false, breg[c],
                                                         (short)0, acc, false, false);
        }
#pragma unroll
        for (int jj = 0; jj < 8; ++jj) {
            float r = acc[jj] + bval;
            if (RELU) r = fmaxf(r, 0.0f);
            size_t idx = (size_t)(mt * 16 + hf * 8 + jj) * N + n;
            if (F16OUT) ((half_t*)Cout)[idx] = (half_t)r;
            else        ((float*)Cout)[idx]  = r;
        }
    }
}

// ---------------------------------------------------------------------------
// Fused attention: one block = (batch b, 16-row tile rt), 8 waves.
//   scores[16,512] = q_tile @ k^T  (WMMA, b128 global loads)    -> LDS f32
//   masked softmax (256 threads, contiguous b128 mask reads,
//                   shuffle reductions)                          -> LDS f16
//   out[16,128] = att @ v, v double-buffer async-staged to LDS in
//   32-agent chunks (copy overlaps WMMA); scores never touch HBM.
// ---------------------------------------------------------------------------
__global__ void attention_wmma(const half_t* __restrict__ q, const half_t* __restrict__ k,
                               const half_t* __restrict__ v, const float* __restrict__ mask,
                               half_t* __restrict__ o) {
    __shared__ __align__(16) float  s_sc[16][NAG];        // 32 KB
    __shared__ __align__(16) half_t s_att[16][NAG];       // 16 KB
    __shared__ __align__(16) half_t s_v[2][32 * HDIM];    // 2 x 8 KB, double buffer

    int b    = blockIdx.x;
    int rt   = blockIdx.y;
    int tid  = threadIdx.x;
    int lane = tid & 31;
    int wave = tid >> 5;

    const half_t* qb = q + ((size_t)b * NAG + rt * 16) * HDIM;
    const half_t* kb = k + (size_t)b * NAG * HDIM;
    const half_t* vb = v + (size_t)b * NAG * HDIM;

    // prefetch this block's 32 KB mask slab while scores WMMAs run
    const float* mbase = mask + ((size_t)b * NAG + rt * 16) * NAG;
    __builtin_prefetch(mbase + (size_t)tid * 32, 0, 1);

    // ---- scores = q @ k^T ----
    for (int ct = wave; ct < (NAG / 16); ct += 8) {
        v8f acc = {};
#pragma unroll
        for (int k0 = 0; k0 < HDIM; k0 += 32) {
            v16h a  = load_A_vec(qb, HDIM, k0, lane);
            v16h bm = load_Bt_vec(kb, HDIM, k0, ct * 16, lane);
            acc = __builtin_amdgcn_wmma_f32_16x16x32_f16(false, a, false, bm,
                                                         (short)0, acc, false, false);
        }
        int col = lane & 15;
        int hf  = lane >> 4;
#pragma unroll
        for (int j = 0; j < 8; ++j)
            s_sc[hf * 8 + j][ct * 16 + col] = acc[j];
    }
    __syncthreads();

    // ---- masked softmax: 16 threads per row, contiguous 32-col span each ----
    {
        int r  = tid >> 4;
        int c0 = tid & 15;
        const float* mrow = mbase + (size_t)r * NAG + c0 * 32;
        const float* srow = &s_sc[r][c0 * 32];
        float vals[32];
        float mx = -3.0e38f;
#pragma unroll
        for (int i = 0; i < 8; ++i) {
            v4f mm = *(const v4f*)(mrow + i * 4);
            v4f ss = *(const v4f*)(srow + i * 4);
#pragma unroll
            for (int t = 0; t < 4; ++t) {
                float sv = ss[t] * mm[t] - 9000000000000000.0f * (1.0f - mm[t]);
                vals[i * 4 + t] = sv;
                mx = fmaxf(mx, sv);
            }
        }
#pragma unroll
        for (int off = 1; off < 16; off <<= 1)
            mx = fmaxf(mx, __shfl_xor(mx, off, 32));
        float sum = 0.0f;
#pragma unroll
        for (int i = 0; i < 32; ++i) {
            float e = __expf(vals[i] - mx);
            vals[i] = e;
            sum += e;
        }
#pragma unroll
        for (int off = 1; off < 16; off <<= 1)
            sum += __shfl_xor(sum, off, 32);
        float inv = 1.0f / sum;
        half_t* arow = &s_att[r][c0 * 32];
#pragma unroll
        for (int i = 0; i < 4; ++i) {
            v8h pk;
#pragma unroll
            for (int t = 0; t < 8; ++t) pk[t] = (half_t)(vals[i * 8 + t] * inv);
            *(v8h*)(arow + i * 8) = pk;
        }
    }
    __syncthreads();

    // ---- out = att @ v, double-buffered async v staging ----
    {
        int col = lane & 15;
        int hf  = lane >> 4;
        v8f acc = {};

        // prologue: stage chunk 0 into buffer 0
#pragma unroll
        for (int i = tid; i < (32 * HDIM) / 8; i += 256)
            async_copy_b128(&s_v[0][i * 8], vb + i * 8);
        async_wait0();
        __syncthreads();

        for (int c = 0; c < NAG / 32; ++c) {
            int cur = c & 1;
            if (c < NAG / 32 - 1) {
                const half_t* src = vb + (size_t)(c + 1) * 32 * HDIM;
#pragma unroll
                for (int i = tid; i < (32 * HDIM) / 8; i += 256)
                    async_copy_b128(&s_v[cur ^ 1][i * 8], src + i * 8);
            }
            v16h a = load_A_vec(&s_att[0][0], NAG, c * 32, lane);
            v16h bm;
#pragma unroll
            for (int i = 0; i < 8; ++i) {
                int kk = hf * 16 + 2 * i;
                bm[2 * i]     = s_v[cur][kk * HDIM + wave * 16 + col];
                bm[2 * i + 1] = s_v[cur][(kk + 1) * HDIM + wave * 16 + col];
            }
            acc = __builtin_amdgcn_wmma_f32_16x16x32_f16(false, a, false, bm,
                                                         (short)0, acc, false, false);
            async_wait0();      // next chunk landed
            __syncthreads();    // all waves done reading s_v[cur]
        }

        half_t* ob = o + ((size_t)b * NAG + rt * 16) * HDIM;
#pragma unroll
        for (int j = 0; j < 8; ++j)
            ob[(size_t)(hf * 8 + j) * HDIM + wave * 16 + col] = (half_t)acc[j];
    }
}

// ---------------------------------------------------------------------------
// Host driver
// ---------------------------------------------------------------------------
extern "C" void kernel_launch(void* const* d_in, const int* in_sizes, int n_in,
                              void* d_out, int out_size, void* d_ws, size_t ws_size,
                              hipStream_t stream) {
    const float* x     = (const float*)d_in[0];
    const float* mask  = (const float*)d_in[1];
    const float* enc_w = (const float*)d_in[2];
    const float* enc_b = (const float*)d_in[3];
    const float* wv    = (const float*)d_in[4];
    const float* bv    = (const float*)d_in[5];
    const float* wk    = (const float*)d_in[6];
    const float* bk    = (const float*)d_in[7];
    const float* wq    = (const float*)d_in[8];
    const float* bq    = (const float*)d_in[9];
    const float* wo    = (const float*)d_in[10];
    const float* bo    = (const float*)d_in[11];
    const float* qw    = (const float*)d_in[12];
    const float* qb    = (const float*)d_in[13];
    float* out = (float*)d_out;

    const size_t ROWS = (size_t)BATCH * NAG;
    const size_t SZ_X = ROWS * DIN;
    const size_t SZ_H = ROWS * HDIM;

    half_t* ws   = (half_t*)d_ws;
    half_t* xh   = ws;
    half_t* hb   = xh + SZ_X;
    half_t* qbuf = hb + SZ_H;
    half_t* kbuf = qbuf + SZ_H;
    half_t* vbuf = kbuf + SZ_H;
    half_t* obuf = vbuf + SZ_H;
    half_t* ench = obuf + SZ_H;
    half_t* wvh  = ench + DIN * HDIM;
    half_t* wkh  = wvh + NPASS * HDIM * HDIM;
    half_t* wqh  = wkh + NPASS * HDIM * HDIM;
    half_t* woh  = wqh + NPASS * HDIM * HDIM;
    half_t* qwh  = woh + NPASS * HDIM * HDIM;

    auto cvt = [&](const float* src, half_t* dst, int n) {
        int n8 = n / 8;
        cvt_f32_f16_v8<<<(n8 + 255) / 256, 256, 0, stream>>>(src, dst, n8);
    };

    cvt(x, xh, (int)SZ_X);
    cvt(enc_w, ench, DIN * HDIM);
    cvt(wv, wvh, NPASS * HDIM * HDIM);
    cvt(wk, wkh, NPASS * HDIM * HDIM);
    cvt(wq, wqh, NPASS * HDIM * HDIM);
    cvt(wo, woh, NPASS * HDIM * HDIM);
    cvt(qw, qwh, HDIM * AOUT);

    const int GB = (int)(ROWS / 128);

    gemm_wmma<DIN, HDIM, true, true><<<GB, 256, 0, stream>>>(xh, ench, enc_b, hb);

    for (int p = 0; p < NPASS; ++p) {
        const half_t* Wv = wvh + (size_t)p * HDIM * HDIM;
        const half_t* Wk = wkh + (size_t)p * HDIM * HDIM;
        const half_t* Wq = wqh + (size_t)p * HDIM * HDIM;
        const half_t* Wo = woh + (size_t)p * HDIM * HDIM;
        gemm_wmma<HDIM, HDIM, true, true><<<GB, 256, 0, stream>>>(hb, Wv, bv + p * HDIM, vbuf);
        gemm_wmma<HDIM, HDIM, true, true><<<GB, 256, 0, stream>>>(hb, Wq, bq + p * HDIM, qbuf);
        gemm_wmma<HDIM, HDIM, true, true><<<GB, 256, 0, stream>>>(hb, Wk, bk + p * HDIM, kbuf);
        attention_wmma<<<dim3(BATCH, NAG / 16), 256, 0, stream>>>(qbuf, kbuf, vbuf, mask, obuf);
        gemm_wmma<HDIM, HDIM, true, true><<<GB, 256, 0, stream>>>(obuf, Wo, bo + p * HDIM, hb);
    }

    gemm_wmma<HDIM, AOUT, false, false><<<GB, 256, 0, stream>>>(hb, qwh, qb, out);
}
